// MoELayer_9165460210286
// MI455X (gfx1250) — matrix operand order repaired and checked by reading, not compiled, not run on hardware
//
#include <hip/hip_runtime.h>
#include <hip/hip_bf16.h>
#include <cstdint>

// Problem constants (match reference)
static constexpr int NTOK = 8 * 4096;   // B*S tokens
static constexpr int HD   = 512;        // hidden dim H
static constexpr int ID   = 1024;       // intermediate dim I
static constexpr int NEXP = 8;          // experts
// Tiling
static constexpr int MT    = 64;        // tokens per block tile
static constexpr int NTILE = 128;       // N rows staged per outer step
static constexpr int KSTEP = 32;        // K per WMMA step (bf16 16x16x32)
static constexpr int WSTR  = 40;        // LDS row stride (elems): 80B rows, 16B aligned
static constexpr int ASTR  = ID + 8;    // act tile row stride (2064B rows, 16B aligned)
static constexpr int NK1   = HD / KSTEP;  // 16
static constexpr int NK2   = ID / KSTEP;  // 32

typedef __attribute__((ext_vector_type(16))) __bf16 v16bf;
typedef __attribute__((ext_vector_type(8)))  float  v8f;

__device__ __forceinline__ unsigned short f32_to_bf16(float f) {
  unsigned u = __float_as_uint(f);
  u += 0x7FFFu + ((u >> 16) & 1u);      // round-to-nearest-even
  return (unsigned short)(u >> 16);
}
__device__ __forceinline__ unsigned pack2(float a, float b) {
  return (unsigned)f32_to_bf16(a) | ((unsigned)f32_to_bf16(b) << 16);
}
__device__ __forceinline__ v8f v8f_zero() {
  v8f z = {0.f,0.f,0.f,0.f,0.f,0.f,0.f,0.f};
  return z;
}

// --- CDNA5 async copy: 16B global -> LDS, tracked by ASYNCcnt -------------
__device__ __forceinline__ void async_cp16(unsigned lds_off, const void* gptr) {
  asm volatile("global_load_async_to_lds_b128 %0, %1, off"
               :: "v"(lds_off), "v"((unsigned long long)gptr) : "memory");
}
__device__ __forceinline__ void wait_async0() {
  asm volatile("s_wait_asynccnt 0x0" ::: "memory");
}
__device__ __forceinline__ unsigned lds_addr(const void* p) {
  return (unsigned)(size_t)p;   // LDS aperture: addr[31:0] = LDS byte offset
}

// A-matrix fragment (16x32 bf16, MxK), per ISA table:
// lanes 0-15: row M=lane, K = {0..7,16..23}; lanes 16-31: same rows, K = {8..15,24..31}
__device__ __forceinline__ v16bf load_fragA(const unsigned short* base, int row0,
                                            int stride, int k0, int lane) {
  const unsigned short* p = base + (row0 + (lane & 15)) * stride + k0 + ((lane >> 4) << 3);
  union { unsigned u[8]; v16bf v; } f;
#pragma unroll
  for (int t = 0; t < 8; ++t) {
    const int k = ((t & 4) << 2) | ((t & 3) << 1);  // 0,2,4,6,16,18,20,22
    f.u[t] = *(const unsigned*)(p + k);
  }
  return f.v;
}

// B-matrix fragment (32x16 bf16, KxN), weight stored row-major [N][K] in LDS:
// lanes 0-15: col N=lane, K = {0..7,16..23}; lanes 16-31: same cols, K = {8..15,24..31}
__device__ __forceinline__ v16bf load_fragB(const unsigned short* base, int n0,
                                            int stride, int k0, int lane) {
  const unsigned short* p = base + (n0 + (lane & 15)) * stride + k0 + ((lane >> 4) << 3);
  union { unsigned u[8]; v16bf v; } f;
#pragma unroll
  for (int t = 0; t < 8; ++t) {
    f.u[t] = *(const unsigned*)(p + (((t & 4) << 2) | ((t & 3) << 1)));  // 0,2,4,6,16,18,20,22
  }
  return f.v;
}

// ---------------------------------------------------------------------------
// Kernel 0: zero output + stats
// ---------------------------------------------------------------------------
__global__ __launch_bounds__(256) void moe_zero(float* __restrict__ out,
                                                float* __restrict__ stats) {
  const long long n4 = (long long)NTOK * HD / 4;
  long long i = (long long)blockIdx.x * blockDim.x + threadIdx.x;
  const long long stride = (long long)gridDim.x * blockDim.x;
  float4* o4 = (float4*)out;
  const float4 z = make_float4(0.f, 0.f, 0.f, 0.f);
  for (; i < n4; i += stride) o4[i] = z;
  if (blockIdx.x == 0 && threadIdx.x < 16) stats[threadIdx.x] = 0.f;
}

// ---------------------------------------------------------------------------
// Kernel 0b: f32 -> packed bf16 pre-convert (one-time per call)
// ---------------------------------------------------------------------------
__global__ __launch_bounds__(256) void convert_bf16(const float* __restrict__ src,
                                                    unsigned* __restrict__ dst,
                                                    long long npair) {
  long long i = (long long)blockIdx.x * blockDim.x + threadIdx.x;
  const long long stride = (long long)gridDim.x * blockDim.x;
  for (; i < npair; i += stride) {
    const float2 v = ((const float2*)src)[i];
    dst[i] = pack2(v.x, v.y);
  }
}

// ---------------------------------------------------------------------------
// Kernel 1: router — logits, top-2 softmax combine weights, aux-loss stats.
// One wave32 per token; 8 tokens per 256-thread block.
// ---------------------------------------------------------------------------
__global__ __launch_bounds__(256) void moe_router(const float* __restrict__ hs,
                                                  const float* __restrict__ w_router,
                                                  float* __restrict__ combine,
                                                  float* __restrict__ stats) {
  __shared__ float wr[NEXP * HD];       // 16 KB
  __shared__ float probs_acc[NEXP];
  __shared__ float usage_acc[NEXP];
  const int tid = threadIdx.x;
  for (int i = tid; i < NEXP * HD; i += 256) wr[i] = w_router[i];
  if (tid < NEXP) { probs_acc[tid] = 0.f; usage_acc[tid] = 0.f; }
  __syncthreads();

  const int wave = tid >> 5, lane = tid & 31;
  const long long token = (long long)blockIdx.x * 8 + wave;
  const float* x = hs + token * HD;

  float logit[NEXP];
#pragma unroll
  for (int e = 0; e < NEXP; ++e) logit[e] = 0.f;
#pragma unroll
  for (int c = 0; c < 4; ++c) {
    const int h = lane * 16 + c * 4;
    const float4 xv = *(const float4*)(x + h);
#pragma unroll
    for (int e = 0; e < NEXP; ++e) {
      const float4 wv = *(const float4*)(&wr[e * HD + h]);
      logit[e] += xv.x * wv.x + xv.y * wv.y + xv.z * wv.z + xv.w * wv.w;
    }
  }
#pragma unroll
  for (int off = 16; off >= 1; off >>= 1)
#pragma unroll
    for (int e = 0; e < NEXP; ++e) logit[e] += __shfl_xor(logit[e], off, 32);

  if (lane == 0) {
    float mx = logit[0];
#pragma unroll
    for (int e = 1; e < NEXP; ++e) mx = fmaxf(mx, logit[e]);
    float p[NEXP], sum = 0.f;
#pragma unroll
    for (int e = 0; e < NEXP; ++e) { p[e] = __expf(logit[e] - mx); sum += p[e]; }
    const float inv = 1.f / sum;
#pragma unroll
    for (int e = 0; e < NEXP; ++e) atomicAdd(&probs_acc[e], p[e] * inv);
    // top-2 (ties: earliest index, matching jax.lax.top_k)
    int i0 = 0;
#pragma unroll
    for (int e = 1; e < NEXP; ++e) if (logit[e] > logit[i0]) i0 = e;
    int i1 = (i0 == 0) ? 1 : 0;
#pragma unroll
    for (int e = 0; e < NEXP; ++e) if (e != i0 && logit[e] > logit[i1]) i1 = e;
    const float m2 = fmaxf(logit[i0], logit[i1]);
    const float e0 = __expf(logit[i0] - m2), e1 = __expf(logit[i1] - m2);
    const float w0 = e0 / (e0 + e1), w1 = e1 / (e0 + e1);
    float cmb[NEXP];
#pragma unroll
    for (int e = 0; e < NEXP; ++e) cmb[e] = 0.f;
    cmb[i0] = w0; cmb[i1] = w1;
#pragma unroll
    for (int e = 0; e < NEXP; ++e) combine[token * NEXP + e] = cmb[e];
    atomicAdd(&usage_acc[i0], 1.f);
    atomicAdd(&usage_acc[i1], 1.f);
  }
  __syncthreads();
  if (tid < NEXP) {
    atomicAdd(&stats[tid],        probs_acc[tid]);
    atomicAdd(&stats[NEXP + tid], usage_acc[tid]);
  }
}

// ---------------------------------------------------------------------------
// Kernel 2: aux-loss finalize (scalar at d_out[B*S*H])
// ---------------------------------------------------------------------------
__global__ void moe_finalize(const float* __restrict__ stats, float* __restrict__ out_aux) {
  if (threadIdx.x == 0 && blockIdx.x == 0) {
    float tot = 0.f;
    for (int e = 0; e < NEXP; ++e) tot += stats[NEXP + e];
    const float denom = fmaxf(tot, 1.f);
    float aux = 0.f;
    for (int e = 0; e < NEXP; ++e)
      aux += (stats[e] / (float)NTOK) * (stats[NEXP + e] / denom);
    out_aux[0] = aux * (float)NEXP;
  }
}

// ---------------------------------------------------------------------------
// Kernel 3: fused expert FFN (bf16 inputs pre-converted in workspace).
// Grid: (NTOK/MT, NEXP). 256 threads = 8 wave32: 2 along M (32 rows) x
// 4 along N (32 cols). Double-buffered async global->LDS staging.
// Phase 1: act[MT x ID] = silu(X Wg^T) * (X Wu^T), kept in LDS as bf16.
// Phase 2: out[MT x HD] += combine * (act Wd^T)   (global fp32 atomics).
// ---------------------------------------------------------------------------
__global__ __launch_bounds__(256) void moe_expert_ffn(const unsigned short* __restrict__ hsb,
                                                      const unsigned short* __restrict__ wgb,
                                                      const unsigned short* __restrict__ wub,
                                                      const unsigned short* __restrict__ wdb,
                                                      const float* __restrict__ combine,
                                                      float* __restrict__ out) {
  __shared__ __align__(16) unsigned short act_lds[MT * ASTR];        // 132,096 B
  __shared__ __align__(16) unsigned short xc_lds[2][MT * WSTR];      // 2 x  5,120 B
  __shared__ __align__(16) unsigned short w0_lds[2][NTILE * WSTR];   // 2 x 10,240 B
  __shared__ __align__(16) unsigned short w1_lds[2][NTILE * WSTR];   // 2 x 10,240 B
  __shared__ float combine_lds[MT];
  __shared__ int any_flag;

  const int e    = blockIdx.y;
  const int m0   = blockIdx.x * MT;
  const int tid  = threadIdx.x;
  const int lane = tid & 31;
  const int wave = tid >> 5;
  const int wm   = wave & 1;    // M group: rows wm*32 .. wm*32+31
  const int wn   = wave >> 1;   // N group: cols wn*32 .. wn*32+31 (within NTILE)

  if (tid == 0) any_flag = 0;
  if (tid < MT) combine_lds[tid] = combine[(long long)(m0 + tid) * NEXP + e];
  __syncthreads();
  if (tid < MT && combine_lds[tid] != 0.f) any_flag = 1;
  __syncthreads();
  if (!any_flag) return;   // no token in this tile routed to expert e

  const unsigned short* wg_e = wgb + (size_t)e * ID * HD;
  const unsigned short* wu_e = wub + (size_t)e * ID * HD;
  const unsigned short* wd_e = wdb + (size_t)e * HD * ID;

  const unsigned xc_base = lds_addr(&xc_lds[0][0]);
  const unsigned w0_base = lds_addr(&w0_lds[0][0]);
  const unsigned w1_base = lds_addr(&w1_lds[0][0]);
  const unsigned XBUFB = MT * WSTR * 2;      // bytes per X buffer
  const unsigned WBUFB = NTILE * WSTR * 2;   // bytes per W buffer

  // per-thread staging coordinates (16B granules; 4 granules per 32-elem row)
  const int srow1 = tid >> 2, sc = tid & 3;          // X:  64 rows * 4  (1/thread)
  const int srow2a = srow1, srow2b = (tid + 256) >> 2; // W: 128 rows * 4 (2/thread)

  // ---------------- Phase 1: gate/up GEMMs + silu fuse ----------------
  for (int n_out = 0; n_out < ID; n_out += NTILE) {
    __syncthreads();   // all prior LDS readers done before re-staging buf 0
    {  // prologue: issue k=0 chunk into buffer 0
      const int k = 0;
      async_cp16(xc_base + srow1 * 80 + sc * 16,
                 hsb + (size_t)(m0 + srow1) * HD + k + sc * 8);
      const size_t ga = (size_t)(n_out + srow2a) * HD + k + sc * 8;
      const size_t gb = (size_t)(n_out + srow2b) * HD + k + sc * 8;
      async_cp16(w0_base + srow2a * 80 + sc * 16, wg_e + ga);
      async_cp16(w0_base + srow2b * 80 + sc * 16, wg_e + gb);
      async_cp16(w1_base + srow2a * 80 + sc * 16, wu_e + ga);
      async_cp16(w1_base + srow2b * 80 + sc * 16, wu_e + gb);
    }
    v8f accg[2][2], accu[2][2];
#pragma unroll
    for (int mi = 0; mi < 2; ++mi)
#pragma unroll
      for (int ji = 0; ji < 2; ++ji) { accg[mi][ji] = v8f_zero(); accu[mi][ji] = v8f_zero(); }

    for (int ks = 0; ks < NK1; ++ks) {
      const int cur = ks & 1;
      wait_async0();       // own async transfers done
      __syncthreads();     // everyone's transfers done -> buf[cur] valid
      if (ks + 1 < NK1) {  // issue next chunk into the other buffer
        const int k = (ks + 1) * KSTEP;
        const unsigned bo = (cur ^ 1) ? 1u : 0u;
        async_cp16(xc_base + bo * XBUFB + srow1 * 80 + sc * 16,
                   hsb + (size_t)(m0 + srow1) * HD + k + sc * 8);
        const size_t ga = (size_t)(n_out + srow2a) * HD + k + sc * 8;
        const size_t gb = (size_t)(n_out + srow2b) * HD + k + sc * 8;
        async_cp16(w0_base + bo * WBUFB + srow2a * 80 + sc * 16, wg_e + ga);
        async_cp16(w0_base + bo * WBUFB + srow2b * 80 + sc * 16, wg_e + gb);
        async_cp16(w1_base + bo * WBUFB + srow2a * 80 + sc * 16, wu_e + ga);
        async_cp16(w1_base + bo * WBUFB + srow2b * 80 + sc * 16, wu_e + gb);
      }
      const v16bf a0 = load_fragA(&xc_lds[cur][0], wm * 32,      WSTR, 0, lane);
      const v16bf a1 = load_fragA(&xc_lds[cur][0], wm * 32 + 16, WSTR, 0, lane);
#pragma unroll
      for (int ji = 0; ji < 2; ++ji) {
        const int nloc = wn * 32 + ji * 16;
        const v16bf bg = load_fragB(&w0_lds[cur][0], nloc, WSTR, 0, lane);
        const v16bf bu = load_fragB(&w1_lds[cur][0], nloc, WSTR, 0, lane);
        accg[0][ji] = __builtin_amdgcn_wmma_f32_16x16x32_bf16(false, a0, false, bg,
                                                              (short)0, accg[0][ji], false, false);
        accg[1][ji] = __builtin_amdgcn_wmma_f32_16x16x32_bf16(false, a1, false, bg,
                                                              (short)0, accg[1][ji], false, false);
        accu[0][ji] = __builtin_amdgcn_wmma_f32_16x16x32_bf16(false, a0, false, bu,
                                                              (short)0, accu[0][ji], false, false);
        accu[1][ji] = __builtin_amdgcn_wmma_f32_16x16x32_bf16(false, a1, false, bu,
                                                              (short)0, accu[1][ji], false, false);
      }
    }

    // epilogue: silu(g)*u -> act_lds (bf16)
    const int half = lane >> 4, nc = lane & 15;
#pragma unroll
    for (int mi = 0; mi < 2; ++mi)
#pragma unroll
      for (int ji = 0; ji < 2; ++ji)
#pragma unroll
        for (int r = 0; r < 8; ++r) {
          const float g = accg[mi][ji][r], u = accu[mi][ji][r];
          const float a = g * __builtin_amdgcn_rcpf(1.f + __expf(-g)) * u;
          const int m = wm * 32 + mi * 16 + r + half * 8;
          const int n = n_out + wn * 32 + ji * 16 + nc;
          act_lds[m * ASTR + n] = f32_to_bf16(a);
        }
  }

  // ---------------- Phase 2: down projection + weighted accumulate ----------------
  for (int n_out = 0; n_out < HD; n_out += NTILE) {
    __syncthreads();
    {  // prologue: Wd chunk k=0 into buffer 0
      async_cp16(w0_base + srow2a * 80 + sc * 16,
                 wd_e + (size_t)(n_out + srow2a) * ID + sc * 8);
      async_cp16(w0_base + srow2b * 80 + sc * 16,
                 wd_e + (size_t)(n_out + srow2b) * ID + sc * 8);
    }
    v8f acc[2][2];
#pragma unroll
    for (int mi = 0; mi < 2; ++mi)
#pragma unroll
      for (int ji = 0; ji < 2; ++ji) acc[mi][ji] = v8f_zero();

    for (int ks = 0; ks < NK2; ++ks) {
      const int cur = ks & 1;
      wait_async0();
      __syncthreads();
      if (ks + 1 < NK2) {
        const int k = (ks + 1) * KSTEP;
        const unsigned bo = (cur ^ 1) ? 1u : 0u;
        async_cp16(w0_base + bo * WBUFB + srow2a * 80 + sc * 16,
                   wd_e + (size_t)(n_out + srow2a) * ID + k + sc * 8);
        async_cp16(w0_base + bo * WBUFB + srow2b * 80 + sc * 16,
                   wd_e + (size_t)(n_out + srow2b) * ID + k + sc * 8);
      }
      const int k = ks * KSTEP;
      const v16bf a0 = load_fragA(act_lds, wm * 32,      ASTR, k, lane);
      const v16bf a1 = load_fragA(act_lds, wm * 32 + 16, ASTR, k, lane);
#pragma unroll
      for (int ji = 0; ji < 2; ++ji) {
        const v16bf b = load_fragB(&w0_lds[cur][0], wn * 32 + ji * 16, WSTR, 0, lane);
        acc[0][ji] = __builtin_amdgcn_wmma_f32_16x16x32_bf16(false, a0, false, b,
                                                             (short)0, acc[0][ji], false, false);
        acc[1][ji] = __builtin_amdgcn_wmma_f32_16x16x32_bf16(false, a1, false, b,
                                                             (short)0, acc[1][ji], false, false);
      }
    }

    const int half = lane >> 4, nc = lane & 15;
#pragma unroll
    for (int mi = 0; mi < 2; ++mi)
#pragma unroll
      for (int ji = 0; ji < 2; ++ji)
#pragma unroll
        for (int r = 0; r < 8; ++r) {
          const int m = wm * 32 + mi * 16 + r + half * 8;
          const int h = n_out + wn * 32 + ji * 16 + nc;
          const float v = acc[mi][ji][r] * combine_lds[m];
          unsafeAtomicAdd(&out[(size_t)(m0 + m) * HD + h], v);
        }
  }
}

// ---------------------------------------------------------------------------
extern "C" void kernel_launch(void* const* d_in, const int* in_sizes, int n_in,
                              void* d_out, int out_size, void* d_ws, size_t ws_size,
                              hipStream_t stream) {
  const float* hs       = (const float*)d_in[0];   // [B,S,H] f32
  const float* w_router = (const float*)d_in[1];   // [E,H]
  const float* w_gate   = (const float*)d_in[2];   // [E,I,H]
  const float* w_up     = (const float*)d_in[3];   // [E,I,H]
  const float* w_down   = (const float*)d_in[4];   // [E,H,I]
  float* out = (float*)d_out;                      // [B,S,H] ++ [aux]

  // Workspace layout (bytes)
  const size_t NHS = (size_t)NTOK * HD;            // 16,777,216 elems
  const size_t NW  = (size_t)NEXP * ID * HD;       //  4,194,304 elems per weight
  char* ws = (char*)d_ws;
  float*          combine = (float*)ws;                                  // 1 MB
  float*          stats   = (float*)(ws + (size_t)NTOK * NEXP * 4);      // 64 B
  unsigned short* hsb     = (unsigned short*)(ws + (size_t)NTOK * NEXP * 4 + 256);
  unsigned short* wgb     = hsb + NHS;
  unsigned short* wub     = wgb + NW;
  unsigned short* wdb     = wub + NW;               // total ~59 MB of ws

  moe_zero<<<2048, 256, 0, stream>>>(out, stats);
  convert_bf16<<<4096, 256, 0, stream>>>(hs,     (unsigned*)hsb, (long long)(NHS / 2));
  convert_bf16<<<2048, 256, 0, stream>>>(w_gate, (unsigned*)wgb, (long long)(NW / 2));
  convert_bf16<<<2048, 256, 0, stream>>>(w_up,   (unsigned*)wub, (long long)(NW / 2));
  convert_bf16<<<2048, 256, 0, stream>>>(w_down, (unsigned*)wdb, (long long)(NW / 2));
  moe_router<<<NTOK / 8, 256, 0, stream>>>(hs, w_router, combine, stats);
  moe_finalize<<<1, 32, 0, stream>>>(stats, out + NHS);
  dim3 grid(NTOK / MT, NEXP);
  moe_expert_ffn<<<grid, 256, 0, stream>>>(hsb, wgb, wub, wdb, combine, out);
}
